// LinearAttention_23295902613522
// MI455X (gfx1250) — matrix operand order repaired and checked by reading, not compiled, not run on hardware
//
#include <hip/hip_runtime.h>
#include <hip/hip_bf16.h>

// Linear attention (B=4, N=4096, D=1024, H=16, hd=64) for MI455X / gfx1250.
// All heavy GEMMs via v_wmma_f32_16x16x32_f16 (fp32 accumulate); f16
// intermediates sized to live in the 192MB L2. GEMM K-tiles are staged
// global->LDS with the Tensor Data Mover (tensor_load_to_lds, TENSORcnt),
// double-buffered so WMMA issue overlaps the DMA.

typedef __attribute__((ext_vector_type(16))) _Float16 v16h;
typedef __attribute__((ext_vector_type(8)))  _Float16 v8h;
typedef __attribute__((ext_vector_type(4)))  _Float16 v4h;
typedef __attribute__((ext_vector_type(8)))  float    v8f;
typedef __attribute__((ext_vector_type(4)))  unsigned u32x4;
typedef __attribute__((ext_vector_type(8)))  int      i32x8;
typedef __attribute__((ext_vector_type(4)))  int      i32x4;

#define EPS_LA 1e-6f

#if defined(__HIP_DEVICE_COMPILE__) && defined(__gfx1250__) && \
    __has_builtin(__builtin_amdgcn_tensor_load_to_lds)
#define USE_TDM 1
#else
#define USE_TDM 0
#endif

#if __has_include(<hip/amd_detail/amd_gfx1250_TDM.h>)
#define TDM_6ARG 1  // therock headers -> 6-arg builtin
#else
#define TDM_6ARG 0  // ROCm 7.2 -> 5-arg builtin
#endif

__device__ __forceinline__ v8f wmma_f16(v16h a, v16h b, v8f c) {
  return __builtin_amdgcn_wmma_f32_16x16x32_f16(
      /*neg_a=*/false, a, /*neg_b=*/false, b,
      /*c_mod=*/(short)0, c, /*reuse_a=*/false, /*reuse_b=*/false);
}

// A-fragment (16x32 f16, row-major in LDS). Lane holds row M = lane&15.
// VGPR0-3: K=(lane<16?0:8)+0..7 ; VGPR4-7: K=16+(lane<16?0:8)+0..7.
// ld must be a multiple of 8 halves (16B alignment per 8-half chunk).
__device__ __forceinline__ v16h frag_a(const _Float16* row0, int ld, int lane) {
  const _Float16* p = row0 + (lane & 15) * ld + ((lane >> 4) << 3);
  v8h lo = *(const v8h*)(p);
  v8h hi = *(const v8h*)(p + 16);
  v16h r;
#pragma unroll
  for (int i = 0; i < 8; ++i) { r[i] = lo[i]; r[i + 8] = hi[i]; }
  return r;
}

// B-fragment (32x16 f16). LDS holds B^T row-major (one row per out column).
// Lane holds column N=lane&15; VGPR0-3: K=(lane<16?0:8)+0..7 ; VGPR4-7: K=16+(lane<16?0:8)+0..7.
__device__ __forceinline__ v16h frag_b(const _Float16* col0, int ld, int lane) {
  const _Float16* p = col0 + (lane & 15) * ld + ((lane >> 4) << 3);
  v8h lo = *(const v8h*)(p);
  v8h hi = *(const v8h*)(p + 16);
  v16h r;
#pragma unroll
  for (int i = 0; i < 8; ++i) { r[i] = lo[i]; r[i + 8] = hi[i]; }
  return r;
}

// Issue a TDM 2-D tile load: 128 rows x 32 halves, row stride `stride_elems`,
// into LDS at byte offset `lds_byte`, padding +16B per 64B row so the LDS
// row stride is 80B (= frag readers' LD of 40 halves).
__device__ __forceinline__ void tdm_load_tile(unsigned lds_byte, const _Float16* g,
                                              int k_rem, int row_rem, int stride_elems) {
#if USE_TDM
  const unsigned long long ga = (unsigned long long)(uintptr_t)g;
  u32x4 g0;
  g0[0] = 1u;                                                  // count=1 (valid)
  g0[1] = lds_byte;                                            // lds_addr
  g0[2] = (unsigned)ga;                                        // global_addr[31:0]
  g0[3] = (unsigned)((ga >> 32) & 0x01FFFFFFu) | 0x80000000u;  // addr[56:32]|type=2
  i32x8 g1;
  // data_size=2B, pad_enable, pad_interval=16dw(code 3), pad_amount=4dw(code 3)
  g1[0] = (1 << 16) | (1 << 20) | (3 << 22) | (3 << 25);
  g1[1] = (int)((k_rem & 0xFFFF) << 16);                           // tdim0[15:0]
  g1[2] = (int)(((unsigned)k_rem >> 16) | ((row_rem & 0xFFFF) << 16)); // tdim0 hi|tdim1 lo
  g1[3] = (int)(((unsigned)row_rem >> 16) | (32u << 16));          // tdim1 hi|tile_dim0=32
  g1[4] = 128;                                                     // tile_dim1=128
  g1[5] = stride_elems;                                            // dim0_stride lo
  g1[6] = 0;
  g1[7] = 0;
  const i32x4 z4 = {0, 0, 0, 0};
#if TDM_6ARG
  const i32x8 z8 = {0, 0, 0, 0, 0, 0, 0, 0};
  __builtin_amdgcn_tensor_load_to_lds(g0, g1, z4, z4, z8, 0);
#else
  __builtin_amdgcn_tensor_load_to_lds(g0, g1, z4, z4, 0);
#endif
#else
  (void)lds_byte; (void)g; (void)k_rem; (void)row_rem; (void)stride_elems;
#endif
}

// fp32 -> f16 conversion (x and the four weight matrices), vectorized.
__global__ __launch_bounds__(256) void cvt_kernel(const float* __restrict__ in,
                                                  _Float16* __restrict__ out, int n4) {
  const int i = blockIdx.x * 256 + threadIdx.x;
  if (i >= n4) return;
  const float4 f = ((const float4*)in)[i];
  v4h h;
  h[0] = (_Float16)f.x; h[1] = (_Float16)f.y;
  h[2] = (_Float16)f.z; h[3] = (_Float16)f.w;
  ((v4h*)out)[i] = h;
}

// Y[M,N] = fmap(A[M,K] @ W[N,K]^T + bias). Tile 128x128, KT=32, 8 waves
// (4 along M x 2 along N -> 2x4 wmma accumulators each). K-tiles staged by
// TDM, double-buffered; wave 0 drives the DMA pipeline.
template <bool FMAP, typename TO>
__global__ __launch_bounds__(256) void gemm_kernel(
    const _Float16* __restrict__ A, const _Float16* __restrict__ W,
    const float* __restrict__ bias, TO* __restrict__ Y,
    int M, int N, int K) {
  constexpr int LD = 40;  // 32 + 8 halves pad -> 80B row stride (TDM pad)
  __shared__ __align__(16) _Float16 sA[2][128 * LD];
  __shared__ __align__(16) _Float16 sB[2][128 * LD];
  const int t = threadIdx.x, lane = t & 31, wid = t >> 5;
  const int wm = (wid & 3) * 32, wn = (wid >> 2) * 64;
  const int m0 = blockIdx.x * 128, n0 = blockIdx.y * 128;
  const int NK = K / 32;

  v8f acc[2][4] = {};

#if USE_TDM
  if (wid == 0) {  // prologue: tile 0 into buffer 0
    tdm_load_tile((unsigned)(uintptr_t)&sA[0][0], A + (size_t)m0 * K, K, M - m0, K);
    tdm_load_tile((unsigned)(uintptr_t)&sB[0][0], W + (size_t)n0 * K, K, N - n0, K);
  }
#endif

  for (int kt = 0; kt < NK; ++kt) {
    const int buf = kt & 1;
    const int k0 = kt * 32;
#if USE_TDM
    if (wid == 0) {
      if (kt + 1 < NK) {  // issue next tile, then wait for current (in-order)
        tdm_load_tile((unsigned)(uintptr_t)&sA[buf ^ 1][0],
                      A + (size_t)m0 * K + k0 + 32, K - k0 - 32, M - m0, K);
        tdm_load_tile((unsigned)(uintptr_t)&sB[buf ^ 1][0],
                      W + (size_t)n0 * K + k0 + 32, K - k0 - 32, N - n0, K);
        __builtin_amdgcn_s_wait_tensorcnt(2);
      } else {
        __builtin_amdgcn_s_wait_tensorcnt(0);
      }
    }
#else
    {  // manual staging fallback (f16 in, f16 LDS)
      const int row = t >> 1, cb = (t & 1) * 16;
      const v8h* ga = (const v8h*)(A + (size_t)(m0 + row) * K + k0 + cb);
      const v8h* gw = (const v8h*)(W + (size_t)(n0 + row) * K + k0 + cb);
      v8h a0 = ga[0], a1 = ga[1], w0 = gw[0], w1 = gw[1];
      v8h* sa = (v8h*)&sA[buf][row * LD + cb];
      v8h* sb = (v8h*)&sB[buf][row * LD + cb];
      sa[0] = a0; sa[1] = a1; sb[0] = w0; sb[1] = w1;
      if (k0 + 32 < K) {
        __builtin_prefetch(A + (size_t)(m0 + row) * K + k0 + 32 + cb, 0, 1);
        __builtin_prefetch(W + (size_t)(n0 + row) * K + k0 + 32 + cb, 0, 1);
      }
    }
#endif
    __syncthreads();  // LDS tile `buf` ready for all waves

    const _Float16* pA = sA[buf];
    const _Float16* pB = sB[buf];
    v16h bfr[4];
#pragma unroll
    for (int ni = 0; ni < 4; ++ni)
      bfr[ni] = frag_b(&pB[(wn + ni * 16) * LD], LD, lane);
#pragma unroll
    for (int mi = 0; mi < 2; ++mi) {
      v16h afr = frag_a(&pA[(wm + mi * 16) * LD], LD, lane);
#pragma unroll
      for (int ni = 0; ni < 4; ++ni)
        acc[mi][ni] = wmma_f16(afr, bfr[ni], acc[mi][ni]);
    }
    __syncthreads();  // all waves done with `buf` before it is re-filled
  }

  // epilogue: bias (+ elu+1 feature map), store
#pragma unroll
  for (int mi = 0; mi < 2; ++mi) {
    const int mb = m0 + wm + mi * 16 + ((lane >> 4) << 3);
#pragma unroll
    for (int ni = 0; ni < 4; ++ni) {
      const int col = n0 + wn + ni * 16 + (lane & 15);
      const float bv = bias[col];
#pragma unroll
      for (int r = 0; r < 8; ++r) {
        float v = acc[mi][ni][r] + bv;
        if constexpr (FMAP) v = (v > 0.f) ? (v + 1.f) : __expf(v);  // elu(v)+1
        Y[(size_t)(mb + r) * N + col] = (TO)v;
      }
    }
  }
}

// ksum[b,h,d] = sum_n k[b,n,h,d] ; one block per (b,h)
__global__ __launch_bounds__(256) void ksum_kernel(
    const _Float16* __restrict__ kh, float* __restrict__ ksum) {
  __shared__ float red[256];
  const int b = blockIdx.x >> 4, h = blockIdx.x & 15;
  const int t = threadIdx.x, d = t & 63, c = t >> 6;
  const _Float16* base = kh + (size_t)b * 4096 * 1024 + (size_t)h * 64 + d;
  float s = 0.f;
  for (int n = c; n < 4096; n += 4) s += (float)base[(size_t)n * 1024];
  red[t] = s;
  __syncthreads();
  if (c == 0)
    ksum[(size_t)blockIdx.x * 64 + d] = red[d] + red[64 + d] + red[128 + d] + red[192 + d];
}

// dinv[b,n,h] = 1 / (q[b,n,h,:] . ksum[b,h,:] + eps)
__global__ __launch_bounds__(256) void dinv_kernel(
    const _Float16* __restrict__ qh, const float* __restrict__ ksum,
    float* __restrict__ dinv) {
  const size_t i = (size_t)blockIdx.x * 256 + threadIdx.x;  // (b*4096+n)*16+h
  const int h = (int)(i & 15);
  const int b = (int)(i >> 16);
  const v8h* q = (const v8h*)(qh + i * 64);
  const float* ks = ksum + ((size_t)b * 16 + h) * 64;
  float s = 0.f;
#pragma unroll
  for (int j = 0; j < 8; ++j) {
    v8h a = q[j];
#pragma unroll
    for (int e = 0; e < 8; ++e) s += (float)a[e] * ks[j * 8 + e];
  }
  dinv[i] = 1.0f / (s + EPS_LA);
}

// ctx[b,h,d,e] = sum_n k[b,n,h,d] * v[b,n,h,e]  (64x64 per head, K=4096).
// One block per (b,h): 8 waves split each 256-wide n-tile, 4x4 wmma accs,
// cross-wave reduce via LDS ds_add_f32.
__global__ __launch_bounds__(256) void context_kernel(
    const _Float16* __restrict__ kh, const _Float16* __restrict__ vh,
    float* __restrict__ ctx) {
  constexpr int LD = 264;  // 256 + 8 halves pad
  __shared__ __align__(16) _Float16 sKT[64 * LD];
  __shared__ __align__(16) _Float16 sVT[64 * LD];
  __shared__ float sC[64 * 64];
  const int bh = blockIdx.x, b = bh >> 4, h = bh & 15;
  const int t = threadIdx.x, lane = t & 31, wid = t >> 5;
  for (int i = t; i < 64 * 64; i += 256) sC[i] = 0.f;

  v8f acc[4][4] = {};
  const size_t rowbase = (size_t)b * 4096 * 1024 + (size_t)h * 64;

  for (int nt = 0; nt < 16; ++nt) {
    __syncthreads();
    {  // stage k/v rows transposed: this thread owns row n = nt*256+t
      const size_t g = rowbase + (size_t)(nt * 256 + t) * 1024;
      const v8h* gk = (const v8h*)(kh + g);
      const v8h* gv = (const v8h*)(vh + g);
      _Float16 kk[64], vv[64];
#pragma unroll
      for (int j = 0; j < 8; ++j) {
        v8h a = gk[j], c = gv[j];
#pragma unroll
        for (int e = 0; e < 8; ++e) { kk[j * 8 + e] = a[e]; vv[j * 8 + e] = c[e]; }
      }
#pragma unroll
      for (int d = 0; d < 64; ++d) {
        sKT[d * LD + t] = kk[d];
        sVT[d * LD + t] = vv[d];
      }
    }
    __syncthreads();
    const int ncol = wid * 32;  // each wave takes a 32-wide K(=n) slice
#pragma unroll
    for (int mi = 0; mi < 4; ++mi) {
      v16h afr = frag_a(&sKT[(mi * 16) * LD + ncol], LD, lane);
#pragma unroll
      for (int ei = 0; ei < 4; ++ei) {
        v16h bfr = frag_b(&sVT[(ei * 16) * LD + ncol], LD, lane);
        acc[mi][ei] = wmma_f16(afr, bfr, acc[mi][ei]);
      }
    }
  }
  __syncthreads();
#pragma unroll
  for (int mi = 0; mi < 4; ++mi)
#pragma unroll
    for (int ei = 0; ei < 4; ++ei) {
      const int e = ei * 16 + (lane & 15);
      const int mb = mi * 16 + ((lane >> 4) << 3);
#pragma unroll
      for (int r = 0; r < 8; ++r)
        atomicAdd(&sC[(mb + r) * 64 + e], acc[mi][ei][r]);  // ds_add_f32
    }
  __syncthreads();
  for (int i = t; i < 64 * 64; i += 256) ctx[(size_t)bh * 4096 + i] = sC[i];
}

// oh[b,n,h,e] = dinv[b,n,h] * sum_d q[b,n,h,d] * ctx[b,h,d,e]
// grid: (B*H, N/128); 8 waves x 16 rows each; K=64 -> 2 wmma steps.
__global__ __launch_bounds__(256) void attn_kernel(
    const _Float16* __restrict__ qh, const float* __restrict__ ctx,
    const float* __restrict__ dinv, _Float16* __restrict__ oh) {
  constexpr int LD = 72;  // 64 + 8 halves pad
  __shared__ __align__(16) _Float16 sQ[128 * LD];
  __shared__ __align__(16) _Float16 sCt[64 * LD];
  const int bh = blockIdx.x, b = bh >> 4, h = bh & 15;
  const int t = threadIdx.x, lane = t & 31, wid = t >> 5;
  const int m0 = blockIdx.y * 128;

  for (int i = t; i < 64 * 64; i += 256) {  // stage ctx^T as f16
    const int d = i >> 6, e = i & 63;
    sCt[e * LD + d] = (_Float16)ctx[(size_t)bh * 4096 + i];
  }
  {  // stage 128 q rows for this head
    const int row = t >> 1, db = (t & 1) * 32;
    const v8h* g = (const v8h*)(qh + ((size_t)(b * 4096 + m0 + row) * 16 + h) * 64 + db);
    v8h* s = (v8h*)&sQ[row * LD + db];
    s[0] = g[0]; s[1] = g[1]; s[2] = g[2]; s[3] = g[3];
  }
  __syncthreads();

  const int wm = wid * 16;
  v8f acc[4] = {};
#pragma unroll
  for (int kk = 0; kk < 64; kk += 32) {
    v16h afr = frag_a(&sQ[wm * LD + kk], LD, lane);
#pragma unroll
    for (int ei = 0; ei < 4; ++ei) {
      v16h bfr = frag_b(&sCt[(ei * 16) * LD + kk], LD, lane);
      acc[ei] = wmma_f16(afr, bfr, acc[ei]);
    }
  }
  const int e0 = lane & 15;
  const int mb = wm + ((lane >> 4) << 3);
#pragma unroll
  for (int r = 0; r < 8; ++r) {
    const int n = m0 + mb + r;
    const float sc = dinv[(size_t)(b * 4096 + n) * 16 + h];
    _Float16* o = oh + ((size_t)(b * 4096 + n) * 16 + h) * 64;
#pragma unroll
    for (int ei = 0; ei < 4; ++ei)
      o[ei * 16 + e0] = (_Float16)(acc[ei][r] * sc);
  }
}

extern "C" void kernel_launch(void* const* d_in, const int* in_sizes, int n_in,
                              void* d_out, int out_size, void* d_ws, size_t ws_size,
                              hipStream_t stream) {
  (void)in_sizes; (void)n_in; (void)out_size; (void)ws_size;
  const float* x  = (const float*)d_in[0];
  const float* Wq = (const float*)d_in[1];
  const float* bq = (const float*)d_in[2];
  const float* Wk = (const float*)d_in[3];
  const float* bk = (const float*)d_in[4];
  const float* Wv = (const float*)d_in[5];
  const float* bv = (const float*)d_in[6];
  const float* Wo = (const float*)d_in[7];
  const float* bo = (const float*)d_in[8];
  float* out = (float*)d_out;

  const int M = 16384, D = 1024;  // M = B*N
  char* w = (char*)d_ws;
  _Float16* xh  = (_Float16*)w; w += (size_t)M * D * 2;  // 32 MB
  _Float16* qh  = (_Float16*)w; w += (size_t)M * D * 2;  // 32 MB
  _Float16* kh  = (_Float16*)w; w += (size_t)M * D * 2;  // 32 MB
  _Float16* vh  = (_Float16*)w; w += (size_t)M * D * 2;  // 32 MB
  _Float16* oh  = (_Float16*)w; w += (size_t)M * D * 2;  // 32 MB
  _Float16* wqh = (_Float16*)w; w += (size_t)D * D * 2;  // 2 MB
  _Float16* wkh = (_Float16*)w; w += (size_t)D * D * 2;  // 2 MB
  _Float16* wvh = (_Float16*)w; w += (size_t)D * D * 2;  // 2 MB
  _Float16* woh = (_Float16*)w; w += (size_t)D * D * 2;  // 2 MB
  float* ksum   = (float*)w;    w += (size_t)4096 * 4;       // 16 KB
  float* ctx    = (float*)w;    w += (size_t)64 * 4096 * 4;  // 1 MB
  float* dinv   = (float*)w;    w += (size_t)M * 16 * 4;     // 1 MB
  // ~170 MB workspace, fully rewritten every launch

  const int n4x = M * D / 4, n4w = D * D / 4;
  cvt_kernel<<<n4x / 256, 256, 0, stream>>>(x,  xh,  n4x);
  cvt_kernel<<<n4w / 256, 256, 0, stream>>>(Wq, wqh, n4w);
  cvt_kernel<<<n4w / 256, 256, 0, stream>>>(Wk, wkh, n4w);
  cvt_kernel<<<n4w / 256, 256, 0, stream>>>(Wv, wvh, n4w);
  cvt_kernel<<<n4w / 256, 256, 0, stream>>>(Wo, woh, n4w);

  const dim3 gg(M / 128, D / 128), gb(256);
  gemm_kernel<true,  _Float16><<<gg, gb, 0, stream>>>(xh, wqh, bq, qh, M, D, D);
  gemm_kernel<true,  _Float16><<<gg, gb, 0, stream>>>(xh, wkh, bk, kh, M, D, D);
  gemm_kernel<false, _Float16><<<gg, gb, 0, stream>>>(xh, wvh, bv, vh, M, D, D);
  ksum_kernel<<<64, 256, 0, stream>>>(kh, ksum);
  context_kernel<<<64, 256, 0, stream>>>(kh, vh, ctx);
  dinv_kernel<<<(M * 16) / 256, 256, 0, stream>>>(qh, ksum, dinv);
  attn_kernel<<<dim3(64, 4096 / 128), 256, 0, stream>>>(qh, ctx, dinv, oh);
  gemm_kernel<false, float><<<gg, gb, 0, stream>>>(oh, woh, bo, out, M, D, D);
}
